// Bidirectional_88038239633468
// MI455X (gfx1250) — compile-verified, matrix operand order checked by reading, not run on hardware
//
#include <hip/hip_runtime.h>
#include <hip/hip_bf16.h>

// ---------------------------------------------------------------------------
// Bidirectional LSTM (B=512,T=256,E=256,U=512) + MLP + softmax for gfx1250.
// Recurrent GEMMs on v_wmma_f32_16x16x32_bf16 (f32 accum), double-buffered
// LDS stages (BK=64), weight tiles delivered by the Tensor Data Mover
// (tensor_load_to_lds + s_wait_tensorcnt), fused gate/cell update per step.
// ---------------------------------------------------------------------------

#define USE_TDM 1

typedef __attribute__((ext_vector_type(16))) __bf16        v16bf;
typedef __attribute__((ext_vector_type(8)))  float         v8f;
typedef __attribute__((ext_vector_type(4)))  unsigned int  u32x4;
typedef __attribute__((ext_vector_type(8)))  unsigned int  u32x8;

static constexpr int Bn = 512;    // batch
static constexpr int Tn = 256;    // time steps
static constexpr int En = 256;    // embed dim
static constexpr int Un = 512;    // hidden units
static constexpr int Vn = 10000;  // vocab

// GEMM tiling for the per-step kernel
static constexpr int BM     = 64;               // batch rows per block
static constexpr int BN     = 128;              // gate columns per block (4 gates x 32 units)
static constexpr int BK     = 64;               // K per stage (2 WMMA K-slices)
static constexpr int LDK    = 80;               // padded LDS k-stride: 160B rows (16B aligned frags)
static constexpr int GST    = 132;              // gates LDS column stride (128 + pad)
static constexpr int NSTAGE = (En + Un) / BK;   // 4 x-stages + 8 h-stages = 12

__device__ __forceinline__ unsigned short f2bf(float f) {
  unsigned int u = __float_as_uint(f);
  unsigned int r = 0x7FFFu + ((u >> 16) & 1u);  // round-to-nearest-even
  return (unsigned short)((u + r) >> 16);
}

union ABFrag {
  uint4 q[2];
  v16bf v;
};

// ---------------------------------------------------------------------------
// Embedding gather + f32 -> bf16 cast
// ---------------------------------------------------------------------------
__global__ __launch_bounds__(256) void embed_cast_kernel(
    const int* __restrict__ sentence, const float* __restrict__ emb,
    unsigned short* __restrict__ xbf) {
  int gid = blockIdx.x * 256 + threadIdx.x;           // < B*T*E
  int e   = gid & (En - 1);
  int bt  = gid >> 8;                                 // En == 256
  int tok = sentence[bt];
  xbf[gid] = f2bf(emb[(long long)tok * En + e]);
}

// ---------------------------------------------------------------------------
// Weight transpose + cast: WT[n][k] = bf16(W[k][n]),  W is [K][N] row-major
// ---------------------------------------------------------------------------
__global__ __launch_bounds__(256) void transpose_cast_kernel(
    const float* __restrict__ W, unsigned short* __restrict__ WT, int K, int N) {
  int gid = blockIdx.x * 256 + threadIdx.x;
  if (gid >= K * N) return;
  int k = gid / N;
  int n = gid - k * N;
  WT[(long long)n * K + k] = f2bf(W[gid]);
}

// ---------------------------------------------------------------------------
// One LSTM time step, both directions, fully fused.
// Block: 64 batch rows x 32 hidden units x all 4 gates. 8 wave32s.
// ---------------------------------------------------------------------------
__global__ __launch_bounds__(256) void lstm_step_kernel(
    const unsigned short* __restrict__ xbf,
    const unsigned short* __restrict__ WxT_fw, const unsigned short* __restrict__ WhT_fw,
    const float* __restrict__ b_fw,
    const unsigned short* __restrict__ WxT_bw, const unsigned short* __restrict__ WhT_bw,
    const float* __restrict__ b_bw,
    const unsigned short* __restrict__ hprev_fw, unsigned short* __restrict__ hnext_fw,
    float* __restrict__ c_fw,
    const unsigned short* __restrict__ hprev_bw, unsigned short* __restrict__ hnext_bw,
    float* __restrict__ c_bw,
    float* __restrict__ hcat, int step) {
  __shared__ __align__(16) unsigned short Atile[2][BM * LDK];
  __shared__ __align__(16) unsigned short Btile[2][BN * LDK];
  __shared__ __align__(16) float          gates[BM * GST];

  const int tid   = threadIdx.x;
  const int lane  = tid & 31;
  const int wave  = tid >> 5;
  const int mBase = blockIdx.x * BM;        // batch row base
  const int u0    = blockIdx.y * 32;        // hidden-unit base
  const int dir   = blockIdx.z;             // 0 = forward, 1 = backward
  const int t     = dir ? (Tn - 1 - step) : step;

  const unsigned short* WxT   = dir ? WxT_bw   : WxT_fw;
  const unsigned short* WhT   = dir ? WhT_bw   : WhT_fw;
  const float*          bias  = dir ? b_bw     : b_fw;
  const unsigned short* hprev = dir ? hprev_bw : hprev_fw;
  unsigned short*       hnext = dir ? hnext_bw : hnext_fw;
  float*                cbuf  = dir ? c_bw     : c_fw;

  // wave tile: 32 rows x 32 cols -> 2x2 WMMA 16x16 accumulators
  const int waveM = (wave & 1) * 32;
  const int waveN = (wave >> 1) * 32;
  const int halfK = (lane >> 4) << 3;       // lanes 16-31 take K+8 per 16-bit layout

  // per-stage A (activation) source: x-phase then h-phase
  auto stageA = [&](int st, const unsigned short*& gA, long long& stride, long long& col) {
    if (st < En / BK) {
      gA = xbf; stride = (long long)Tn * En; col = (long long)t * En + st * BK;
    } else {
      gA = hprev; stride = Un; col = (long long)(st - En / BK) * BK;
    }
  };

#if USE_TDM
  // Issue one 3D-tile TDM load for the weight tile of stage `st` into Btile[buf]:
  // tile = 64 K-elems x 32 rows x 4 gate-chunks; LDS pad rebuilds the 160B row
  // stride (32 DWORDs data -> pad_interval=4, 8 DWORDs pad -> pad_amount=7).
  auto issueTDM = [&](int st, int buf) {
    const unsigned short* wt;
    unsigned kdim, kloc;
    if (st < En / BK) { wt = WxT; kdim = En; kloc = (unsigned)(st * BK); }
    else              { wt = WhT; kdim = Un; kloc = (unsigned)((st - En / BK) * BK); }
    unsigned long long ga  = (unsigned long long)wt +
                             ((unsigned long long)u0 * kdim + kloc) * 2ull;
    unsigned ldsOff = (unsigned)(unsigned long long)(const void*)&Btile[buf][0];
    unsigned d1s    = 512u * kdim;          // tensor_dim1_stride (elements)

    u32x4 g0;                               // D# group 0
    g0.x = 1u;                              // count=1, user descriptor
    g0.y = ldsOff;                          // lds_addr
    g0.z = (unsigned)ga;                    // global_addr[31:0]
    g0.w = (unsigned)(ga >> 32) | 0x80000000u;  // global_addr[56:32] | type=2

    u32x8 g1;                               // D# group 1
    g1[0] = (1u << 16) | (1u << 20) | (4u << 22) | (7u << 25);  // 2B elems, pad on
    g1[1] = kdim << 16;                     // tensor_dim0[15:0]
    g1[2] = 2048u << 16;                    // tensor_dim1[15:0] = 4U rows
    g1[3] = 64u << 16;                      // tile_dim0 = 64
    g1[4] = 32u | (4u << 16);               // tile_dim1 = 32, tile_dim2 = 4
    g1[5] = kdim;                           // tensor_dim0_stride[31:0]
    g1[6] = (d1s & 0xffffu) << 16;          // tensor_dim1_stride[15:0]
    g1[7] = d1s >> 16;                      // tensor_dim1_stride[47:16]

    u32x4 g2;                               // D# group 2 (3D extent)
    g2.x = 4u; g2.y = 0u; g2.z = 0u; g2.w = 0u;
    u32x4 g3;                               // D# group 3 (unused dims)
    g3.x = 0u; g3.y = 0u; g3.z = 0u; g3.w = 0u;

    asm volatile("tensor_load_to_lds %0, %1, %2, %3"
                 :: "s"(g0), "s"(g1), "s"(g2), "s"(g3)
                 : "memory");
  };
#else
  auto loadBManual = [&](int st, int buf) {
    const unsigned short* wt;
    long long kdim, kloc;
    if (st < En / BK) { wt = WxT; kdim = En; kloc = st * BK; }
    else              { wt = WhT; kdim = Un; kloc = (st - En / BK) * BK; }
#pragma unroll
    for (int j = 0; j < 4; ++j) {
      int c = tid + j * 256;
      int nl = c >> 3, cc = c & 7;
      int ngl = ((nl >> 5) * Un) + u0 + (nl & 31);
      uint4 d = *(const uint4*)(wt + (long long)ngl * kdim + kloc + cc * 8);
      *(uint4*)&Btile[buf][nl * LDK + cc * 8] = d;
    }
  };
#endif

  v8f acc[2][2];
#pragma unroll
  for (int i = 0; i < 2; ++i)
#pragma unroll
    for (int j = 0; j < 2; ++j)
      acc[i][j] = (v8f){0.f, 0.f, 0.f, 0.f, 0.f, 0.f, 0.f, 0.f};

  // ---- prologue: stage 0 into buffer 0
  {
    const unsigned short* gA; long long as, ac;
    stageA(0, gA, as, ac);
#if USE_TDM
    if (wave == 0) issueTDM(0, 0);
#else
    loadBManual(0, 0);
#endif
#pragma unroll
    for (int j = 0; j < 2; ++j) {          // A: 64 rows x 64 el = 512 x 16B chunks
      int c = tid + j * 256;
      int row = c >> 3, cc = c & 7;
      uint4 d = *(const uint4*)(gA + (long long)(mBase + row) * as + ac + cc * 8);
      *(uint4*)&Atile[0][row * LDK + cc * 8] = d;
    }
#if USE_TDM
    if (wave == 0) __builtin_amdgcn_s_wait_tensorcnt(0);
#endif
    __syncthreads();
  }

  // ---- main pipeline: prefetch stage ks+1 while computing stage ks
  for (int ks = 0; ks < NSTAGE; ++ks) {
    const int cur = ks & 1, nxt = cur ^ 1;
    const bool hasNext = (ks + 1 < NSTAGE);

    uint4 pf[2];
    if (hasNext) {
      const unsigned short* gA; long long as, ac;
      stageA(ks + 1, gA, as, ac);
#if USE_TDM
      if (wave == 0) issueTDM(ks + 1, nxt);
#else
      loadBManual(ks + 1, nxt);
#endif
#pragma unroll
      for (int j = 0; j < 2; ++j) {
        int c = tid + j * 256;
        int row = c >> 3, cc = c & 7;
        pf[j] = *(const uint4*)(gA + (long long)(mBase + row) * as + ac + cc * 8);
      }
    }

    // compute on the current buffer: 2 K-slices x (2x2) WMMA tiles
#pragma unroll
    for (int ksl = 0; ksl < 2; ++ksl) {
      ABFrag af[2], bfr[2];
      const int kof = ksl * 32 + halfK;
#pragma unroll
      for (int mt = 0; mt < 2; ++mt) {
        int off = (waveM + mt * 16 + (lane & 15)) * LDK + kof;
        af[mt].q[0] = *(const uint4*)&Atile[cur][off];
        af[mt].q[1] = *(const uint4*)&Atile[cur][off + 16];
      }
#pragma unroll
      for (int nt = 0; nt < 2; ++nt) {
        int off = (waveN + nt * 16 + (lane & 15)) * LDK + kof;
        bfr[nt].q[0] = *(const uint4*)&Btile[cur][off];
        bfr[nt].q[1] = *(const uint4*)&Btile[cur][off + 16];
      }
#pragma unroll
      for (int mt = 0; mt < 2; ++mt)
#pragma unroll
        for (int nt = 0; nt < 2; ++nt)
          acc[mt][nt] = __builtin_amdgcn_wmma_f32_16x16x32_bf16(
              false, af[mt].v, false, bfr[nt].v, (short)0, acc[mt][nt], false, false);
    }

    if (hasNext) {
#pragma unroll
      for (int j = 0; j < 2; ++j) {
        int c = tid + j * 256;
        int row = c >> 3, cc = c & 7;
        *(uint4*)&Atile[nxt][row * LDK + cc * 8] = pf[j];
      }
#if USE_TDM
      if (wave == 0) __builtin_amdgcn_s_wait_tensorcnt(0);
#endif
    }
    __syncthreads();
  }

  // ---- spill accumulators to LDS so each thread can see all 4 gates of a unit
#pragma unroll
  for (int mt = 0; mt < 2; ++mt)
#pragma unroll
    for (int nt = 0; nt < 2; ++nt)
#pragma unroll
      for (int v = 0; v < 8; ++v) {
        int row = waveM + mt * 16 + ((lane >> 4) << 3) + v;
        int col = waveN + nt * 16 + (lane & 15);
        gates[row * GST + col] = acc[mt][nt][v];
      }
  __syncthreads();

  // ---- LSTM cell update: 64 rows x 32 units, 8 elements per thread
  for (int e = tid; e < BM * 32; e += 256) {
    int row  = e >> 5;
    int u    = e & 31;
    int gcol = u0 + u;
    float ig = gates[row * GST + u]      + bias[0 * Un + gcol];
    float fg = gates[row * GST + 32 + u] + bias[1 * Un + gcol];
    float gg = gates[row * GST + 64 + u] + bias[2 * Un + gcol];
    float og = gates[row * GST + 96 + u] + bias[3 * Un + gcol];
    float i_ = 1.f / (1.f + __expf(-ig));
    float f_ = 1.f / (1.f + __expf(-fg));
    float o_ = 1.f / (1.f + __expf(-og));
    float g_ = tanhf(gg);
    long long ci = (long long)(mBase + row) * Un + gcol;
    float c_ = f_ * cbuf[ci] + i_ * g_;
    cbuf[ci] = c_;
    float h_ = o_ * tanhf(c_);
    hnext[ci] = f2bf(h_);
    hcat[(long long)(mBase + row) * (2 * Un) + dir * Un + gcol] = h_;
  }
}

// ---------------------------------------------------------------------------
// hidden = relu(hcat @ W1 + b1)   [512,1024] x [1024,64]
// ---------------------------------------------------------------------------
__global__ __launch_bounds__(64) void mlp_hidden_kernel(
    const float* __restrict__ hcat, const float* __restrict__ W1,
    const float* __restrict__ b1, float* __restrict__ hidden) {
  __shared__ float sh[2 * Un];
  int b = blockIdx.x, j = threadIdx.x;
  for (int k = j; k < 2 * Un; k += 64) sh[k] = hcat[(long long)b * (2 * Un) + k];
  __syncthreads();
  float acc = b1[j];
  for (int k = 0; k < 2 * Un; ++k) acc += sh[k] * W1[k * 64 + j];
  hidden[b * 64 + j] = fmaxf(acc, 0.f);
}

// ---------------------------------------------------------------------------
// out = softmax(hidden @ W2 + b2)   [512,64] x [64,10000], one block per row
// ---------------------------------------------------------------------------
__global__ __launch_bounds__(256) void mlp_softmax_kernel(
    const float* __restrict__ hidden, const float* __restrict__ W2,
    const float* __restrict__ b2, float* __restrict__ out) {
  __shared__ float sh[64];
  __shared__ float red[256];
  int b = blockIdx.x, tid = threadIdx.x;
  if (tid < 64) sh[tid] = hidden[b * 64 + tid];
  __syncthreads();

  float lmax = -3.4e38f;
  for (int v = tid; v < Vn; v += 256) {
    float acc = b2[v];
#pragma unroll 8
    for (int k = 0; k < 64; ++k) acc += sh[k] * W2[k * Vn + v];
    out[(long long)b * Vn + v] = acc;
    lmax = fmaxf(lmax, acc);
  }
  red[tid] = lmax;
  __syncthreads();
  for (int s = 128; s > 0; s >>= 1) {
    if (tid < s) red[tid] = fmaxf(red[tid], red[tid + s]);
    __syncthreads();
  }
  float m = red[0];
  __syncthreads();

  float lsum = 0.f;
  for (int v = tid; v < Vn; v += 256) {
    float e = __expf(out[(long long)b * Vn + v] - m);
    out[(long long)b * Vn + v] = e;
    lsum += e;
  }
  red[tid] = lsum;
  __syncthreads();
  for (int s = 128; s > 0; s >>= 1) {
    if (tid < s) red[tid] += red[tid + s];
    __syncthreads();
  }
  float inv = 1.f / red[0];
  for (int v = tid; v < Vn; v += 256) out[(long long)b * Vn + v] *= inv;
}

// ---------------------------------------------------------------------------
extern "C" void kernel_launch(void* const* d_in, const int* in_sizes, int n_in,
                              void* d_out, int out_size, void* d_ws, size_t ws_size,
                              hipStream_t stream) {
  const int*   sentence = (const int*)d_in[0];
  const float* emb      = (const float*)d_in[1];
  const float* Wx_fw    = (const float*)d_in[2];
  const float* Wh_fw    = (const float*)d_in[3];
  const float* b_fw     = (const float*)d_in[4];
  const float* Wx_bw    = (const float*)d_in[5];
  const float* Wh_bw    = (const float*)d_in[6];
  const float* b_bw     = (const float*)d_in[7];
  const float* W1       = (const float*)d_in[8];
  const float* b1       = (const float*)d_in[9];
  const float* W2       = (const float*)d_in[10];
  const float* b2       = (const float*)d_in[11];
  float*       out      = (float*)d_out;

  // ---- workspace carve (all 256B aligned) ----
  char* p = (char*)d_ws;
  auto take = [&](size_t bytes) {
    void* r = (void*)p;
    p += (bytes + 255) & ~size_t(255);
    return r;
  };
  unsigned short* xbf  = (unsigned short*)take((size_t)Bn * Tn * En * 2);  // 67 MB
  unsigned short* WxTf = (unsigned short*)take((size_t)4 * Un * En * 2);
  unsigned short* WhTf = (unsigned short*)take((size_t)4 * Un * Un * 2);
  unsigned short* WxTb = (unsigned short*)take((size_t)4 * Un * En * 2);
  unsigned short* WhTb = (unsigned short*)take((size_t)4 * Un * Un * 2);
  unsigned short* hA_f = (unsigned short*)take((size_t)Bn * Un * 2);
  unsigned short* hB_f = (unsigned short*)take((size_t)Bn * Un * 2);
  unsigned short* hA_b = (unsigned short*)take((size_t)Bn * Un * 2);
  unsigned short* hB_b = (unsigned short*)take((size_t)Bn * Un * 2);
  float*          c_f  = (float*)take((size_t)Bn * Un * 4);
  float*          c_b  = (float*)take((size_t)Bn * Un * 4);
  float*          hcat = (float*)take((size_t)Bn * 2 * Un * 4);
  float*          hid  = (float*)take((size_t)Bn * 64 * 4);

  // zero initial states (stream-ordered; graph-capturable)
  hipMemsetAsync(hA_f, 0, (size_t)Bn * Un * 2, stream);
  hipMemsetAsync(hB_f, 0, (size_t)Bn * Un * 2, stream);
  hipMemsetAsync(hA_b, 0, (size_t)Bn * Un * 2, stream);
  hipMemsetAsync(hB_b, 0, (size_t)Bn * Un * 2, stream);
  hipMemsetAsync(c_f, 0, (size_t)Bn * Un * 4, stream);
  hipMemsetAsync(c_b, 0, (size_t)Bn * Un * 4, stream);

  // one-time prep
  embed_cast_kernel<<<(Bn * Tn * En) / 256, 256, 0, stream>>>(sentence, emb, xbf);
  transpose_cast_kernel<<<(En * 4 * Un + 255) / 256, 256, 0, stream>>>(Wx_fw, WxTf, En, 4 * Un);
  transpose_cast_kernel<<<(Un * 4 * Un + 255) / 256, 256, 0, stream>>>(Wh_fw, WhTf, Un, 4 * Un);
  transpose_cast_kernel<<<(En * 4 * Un + 255) / 256, 256, 0, stream>>>(Wx_bw, WxTb, En, 4 * Un);
  transpose_cast_kernel<<<(Un * 4 * Un + 255) / 256, 256, 0, stream>>>(Wh_bw, WhTb, Un, 4 * Un);

  // recurrent scan: fw consumes t = 0..T-1, bw consumes t = T-1..0, h double-buffered
  dim3 grid(Bn / BM, Un / 32, 2);
  for (int s = 0; s < Tn; ++s) {
    const unsigned short* hp_f = (s & 1) ? hB_f : hA_f;
    unsigned short*       hn_f = (s & 1) ? hA_f : hB_f;
    const unsigned short* hp_b = (s & 1) ? hB_b : hA_b;
    unsigned short*       hn_b = (s & 1) ? hA_b : hB_b;
    lstm_step_kernel<<<grid, 256, 0, stream>>>(
        xbf, WxTf, WhTf, b_fw, WxTb, WhTb, b_bw,
        hp_f, hn_f, c_f, hp_b, hn_b, c_b, hcat, s);
  }

  // classifier head
  mlp_hidden_kernel<<<Bn, 64, 0, stream>>>(hcat, W1, b1, hid);
  mlp_softmax_kernel<<<Bn, 256, 0, stream>>>(hid, W2, b2, out);
}
